// GraphTransformerPooling_40587440947795
// MI455X (gfx1250) — matrix operand order repaired and checked by reading, not compiled, not run on hardware
//
#include <hip/hip_runtime.h>
#include <hip/hip_bf16.h>

// Problem constants (match reference)
static constexpr int kN    = 131072;
static constexpr int kG    = 256;
static constexpr int kE    = 384;
static constexpr int kH    = 6;
static constexpr int kDH   = 64;
static constexpr int kOUT  = 256;
static constexpr int kXS   = 392;   // row-major LDS stride (bf16); 392*2=784 is 16B-aligned
static constexpr int kXT   = 80;    // feature-major LDS stride (bf16); 80*2=160 is 16B-aligned

typedef __attribute__((ext_vector_type(16))) __bf16 v16bf;
typedef __attribute__((ext_vector_type(8)))  float  v8f;

// ---- helpers -------------------------------------------------------------

// float -> bf16 round-to-nearest-even (bit trick; portable across toolchains)
static __device__ __forceinline__ unsigned short bfbits(float x) {
  unsigned u = __builtin_bit_cast(unsigned, x);
  u += 0x7fffu + ((u >> 16) & 1u);
  return (unsigned short)(u >> 16);
}
static __device__ __forceinline__ __bf16 tobf(float x) {
  unsigned short s = bfbits(x);
  return __builtin_bit_cast(__bf16, s);
}

// Assemble a v16bf fragment from two aligned 16B words
struct bfpack { uint4 lo, hi; };
static __device__ __forceinline__ v16bf pack16(uint4 lo, uint4 hi) {
  bfpack t{lo, hi};
  return __builtin_bit_cast(v16bf, t);
}

// K-offset inside a 16-bit A fragment (16x32), per CDNA5 ISA 7.12.2:
//   lanes 0-15 : e0-7 -> K=0..7,  e8-15 -> K=16..23
//   lanes 16-31: e0-7 -> K=8..15, e8-15 -> K=24..31
static __device__ __forceinline__ int kOffA(int e, int hv) {
  int p = e >> 1;
  return ((p < 4) ? 0 : 16) + hv * 8 + ((p & 3) << 1) + (e & 1);
}

static __device__ __forceinline__ v8f zero8() {
  v8f z;
#pragma unroll
  for (int i = 0; i < 8; ++i) z[i] = 0.f;
  return z;
}

static __device__ __forceinline__ v8f wmma_bf16(v16bf a, v16bf b, v8f c) {
  return __builtin_amdgcn_wmma_f32_16x16x32_bf16(false, a, false, b, (short)0, c,
                                                 false, false);
}

// ---- kernel 1: offsets scan + q projection + qk_h = Wk_h^T q_h (bf16, head-major [16][E]) ----
__global__ void gtp_setup(const float* __restrict__ query,
                          const float* __restrict__ Wq,
                          const float* __restrict__ bq,
                          const float* __restrict__ Wk,
                          const int*   __restrict__ sizes,
                          int*            __restrict__ offsets,
                          unsigned short* __restrict__ qkT /* [16][kE] bf16, heads 6..15 zero */) {
  __shared__ float qs[kE];
  const int tid = threadIdx.x;  // kE = 384 threads
  if (tid == 0) {
    int acc = 0;
    for (int g = 0; g < kG; ++g) { offsets[g] = acc; acc += sizes[g]; }
  }
  // q = (Wq @ query + bq) * rsqrt(DH)
  {
    float s = 0.f;
    const float* wr = Wq + (size_t)tid * kE;
    for (int e = 0; e < kE; ++e) s += wr[e] * query[e];
    qs[tid] = (s + bq[tid]) * 0.125f;  // 1/sqrt(64)
  }
  __syncthreads();
  // qkT[h][e] = sum_d qs[h*64+d] * Wk[h*64+d][e]   (bk shifts all scores of a head
  // by a constant within each segment -> cancels in softmax, dropped)
  const int e = tid;
  for (int h = 0; h < 16; ++h) {
    float s = 0.f;
    if (h < kH) {
      for (int d = 0; d < kDH; ++d) s += qs[h * kDH + d] * Wk[(size_t)(h * kDH + d) * kE + e];
    }
    qkT[h * kE + e] = bfbits(s);
  }
}

// ---- kernel 2: single-pass online-softmax attention per segment ----
// Emits xbar[g][h][e] = sum_i softmax_i * node_feat[i][e]
__global__ void __launch_bounds__(128)
gtp_attn(const float* __restrict__ node_feat,
         const int*   __restrict__ sizes,
         const int*   __restrict__ offsets,
         const unsigned short* __restrict__ qkT,
         float* __restrict__ xbar /* [kG][kH][kE] */) {
  __shared__ __bf16 Xs [64 * kXS];    // chunk, row-major      (~50 KB) -> score A fragments
  __shared__ __bf16 XsT[kE * kXT];    // chunk, feature-major  (~60 KB) -> accum  B fragments
  __shared__ float  Sc[16][64];       // chunk scores -> P; rows 6..15 stay zero
  __shared__ float  mrun[kH], lrun[kH], scL[kH];

  const int g    = blockIdx.x;
  const int n    = sizes[g];
  const int off  = offsets[g];
  const int tid  = threadIdx.x;
  const int wave = tid >> 5;
  const int lane = tid & 31;
  const int m    = lane & 15;  // A-row / B-col / D-col within fragment
  const int hv   = lane >> 4;  // lane half

  // init: zero padded P rows once; running stats
  for (int i = tid; i < 16 * 64; i += 128) Sc[i >> 6][i & 63] = 0.f;
  if (tid < kH) { mrun[tid] = -3.0e38f; lrun[tid] = 0.f; }

  v8f accv[6];                         // per-wave: 6 feature tiles of 16
#pragma unroll
  for (int t = 0; t < 6; ++t) accv[t] = zero8();

  const int nCh = (n + 63) >> 6;       // block-uniform
  for (int c = 0; c < nCh; ++c) {
    __syncthreads();  // (A) previous chunk's WMMA/lrun reads of Xs/XsT/Sc done

    // ---- stage 64 rows as bf16 into LDS in both orientations + prefetch next ----
    for (int t = tid; t < 64 * kE; t += 128) {
      int r = t / kE, e = t - r * kE;
      int gr = off + c * 64 + r;
      gr = gr < (kN - 1) ? gr : (kN - 1);
      __bf16 v = tobf(node_feat[(size_t)gr * kE + e]);
      Xs [r * kXS + e] = v;            // row-major (K = feature contiguous)
      XsT[e * kXT + r] = v;            // feature-major (K = node contiguous)
      if ((tid & 31) == 0) {           // one lane per 128B line
        int gp = off + (c + 1) * 64 + r;
        gp = gp < (kN - 1) ? gp : (kN - 1);
        __builtin_prefetch(node_feat + (size_t)gp * kE + e, 0, 0);
      }
    }
    __syncthreads();  // (B) Xs/XsT ready

    // ---- scores: wave w computes 16-row tile w of this chunk ----
    {
      v8f acc = zero8();
      const __bf16* xbase = &Xs[(wave * 16 + m) * kXS];
      const unsigned short* qbase = qkT + (size_t)m * kE;
      for (int kb = 0; kb < kE; kb += 32) {
        // A: two contiguous 16B runs per lane (see kOffA layout)
        const uint4* xp = (const uint4*)(xbase + kb + hv * 8);
        v16bf a = pack16(xp[0], xp[2]);            // +0 and +16 elements (=+32B)
        const uint4* qp = (const uint4*)(qbase + kb + hv * 8);
        v16bf b = pack16(qp[0], qp[2]);            // same K layout as A (kOffA)
        acc = wmma_bf16(a, b, acc);
      }
#pragma unroll
      for (int r = 0; r < 8; ++r) {
        int lrow = wave * 16 + r + hv * 8;         // row within chunk
        if (m < kH) Sc[m][lrow] = (c * 64 + lrow < n) ? acc[r] : -1e30f;
      }
    }
    __syncthreads();  // (C) Sc ready

    // ---- online softmax stats (6 threads; 64 values each) ----
    if (tid < kH) {
      float cmax = -3.0e38f;
      for (int i = 0; i < 64; ++i) cmax = fmaxf(cmax, Sc[tid][i]);
      float mnew = fmaxf(mrun[tid], cmax);
      scL[tid]   = __expf(mrun[tid] - mnew);       // rescale for old acc/sum
      mrun[tid]  = mnew;
    }
    __syncthreads();  // (D) scL, mrun ready

    // exponentiate chunk (rows >= n were -1e30 -> exp ~ 0)
    for (int idx = tid; idx < kH * 64; idx += 128) {
      int h = idx >> 6, i = idx & 63;
      Sc[h][i] = __expf(Sc[h][i] - mrun[h]);
    }
    // rescale running accumulators (register op, overlaps with exp)
#pragma unroll
    for (int nt = 0; nt < 6; ++nt)
#pragma unroll
      for (int r = 0; r < 8; ++r) {
        int h = r + hv * 8;
        if (h < kH) accv[nt][r] *= scL[h];
      }
    __syncthreads();  // (E) P ready

    if (tid < kH) {
      float s = 0.f;
      for (int i = 0; i < 64; ++i) s += Sc[tid][i];
      lrun[tid] = lrun[tid] * scL[tid] + s;
    }

    // ---- accumulate: accv += P[16 x 64] @ X[64 x E], each wave 6 feature tiles ----
#pragma unroll
    for (int ks = 0; ks < 2; ++ks) {
      v16bf a;
#pragma unroll
      for (int e2 = 0; e2 < 16; ++e2)
        a[e2] = tobf(Sc[m][ks * 32 + kOffA(e2, hv)]);  // rows 6..15 are zero
#pragma unroll
      for (int nt = 0; nt < 6; ++nt) {
        const int featBase = (wave * 6 + nt) * 16;
        // B from feature-major staging: two 8-node K runs per lane (kOffA layout)
        const uint4* bp_ = (const uint4*)(&XsT[(featBase + m) * kXT + ks * 32 + hv * 8]);
        v16bf b = pack16(bp_[0], bp_[2]);
        accv[nt] = wmma_bf16(a, b, accv[nt]);
      }
    }
  }
  __syncthreads();  // final lrun visible

  // write xbar scaled by 1/l (softmax normalization)
#pragma unroll
  for (int nt = 0; nt < 6; ++nt) {
    const int feat = (wave * 6 + nt) * 16 + m;
#pragma unroll
    for (int r = 0; r < 8; ++r) {
      int h = r + hv * 8;
      if (h < kH)
        xbar[((size_t)g * kH + h) * kE + feat] = accv[nt][r] / lrun[h];
    }
  }
}

// ---- kernel 3: pooled[g][j] = Wv[j,:] . xbar[g][j/64][:] + bv[j] ----
// (sum of attention weights is 1 -> bv passes through with coefficient 1)
__global__ void gtp_wv(const float* __restrict__ Wv,
                       const float* __restrict__ bv,
                       const float* __restrict__ xbar,
                       float* __restrict__ pooled) {
  const int g = blockIdx.x, j = threadIdx.x;   // 384 threads
  const int h = j >> 6;                        // j / 64
  const float* xb = xbar + ((size_t)g * kH + h) * kE;
  const float* wr = Wv + (size_t)j * kE;
  float s = bv[j];
  for (int e = 0; e < kE; ++e) s += wr[e] * xb[e];
  pooled[(size_t)g * kE + j] = s;
}

// ---- kernel 4: generic C[M,Nc] = A[M,K] @ W[Nc,K]^T + bias, one wave per 16x16 tile ----
__global__ void __launch_bounds__(32)
gtp_gemm(const float* __restrict__ A,
         const float* __restrict__ W,
         const float* __restrict__ bias,
         float* __restrict__ C,
         int Nc, int K) {
  const int lane  = threadIdx.x & 31;
  const int m     = lane & 15;
  const int hv    = lane >> 4;
  const int nBase = blockIdx.x * 16;
  const int mBase = blockIdx.y * 16;
  v8f acc = zero8();
  for (int kb = 0; kb < K; kb += 32) {
    v16bf a, b;
    const float* ar = A + (size_t)(mBase + m) * K + kb;
#pragma unroll
    for (int e2 = 0; e2 < 16; ++e2) a[e2] = tobf(ar[kOffA(e2, hv)]);
    const float* wr = W + (size_t)(nBase + m) * K + kb;
#pragma unroll
    for (int e2 = 0; e2 < 16; ++e2) b[e2] = tobf(wr[kOffA(e2, hv)]);
    acc = wmma_bf16(a, b, acc);
  }
  const float bcol = bias[nBase + m];
#pragma unroll
  for (int r = 0; r < 8; ++r)
    C[(size_t)(mBase + r + hv * 8) * Nc + nBase + m] = acc[r] + bcol;
}

// ---- launch --------------------------------------------------------------
extern "C" void kernel_launch(void* const* d_in, const int* in_sizes, int n_in,
                              void* d_out, int out_size, void* d_ws, size_t ws_size,
                              hipStream_t stream) {
  (void)in_sizes; (void)n_in; (void)out_size; (void)ws_size;
  const float* node_feat = (const float*)d_in[0];
  const int*   sizes     = (const int*)d_in[1];
  const float* query     = (const float*)d_in[2];
  const float* Wq        = (const float*)d_in[3];
  const float* bq        = (const float*)d_in[4];
  const float* Wk        = (const float*)d_in[5];
  // d_in[6] = bk: per-head constant score shift -> cancels in softmax, unused
  const float* Wv        = (const float*)d_in[7];
  const float* bv        = (const float*)d_in[8];
  const float* Wo        = (const float*)d_in[9];
  const float* bo        = (const float*)d_in[10];
  const float* Wp        = (const float*)d_in[11];
  const float* bp        = (const float*)d_in[12];
  float* out = (float*)d_out;

  // workspace carve-up (all 16B aligned)
  char* ws = (char*)d_ws;
  int*            offsets = (int*)ws;                               //   1 KB
  unsigned short* qkT     = (unsigned short*)(ws + 1024);           //  12 KB
  float* xbar   = (float*)(ws + 16384);                             // kG*kH*kE*4 = 2304 KB
  float* pooled = (float*)(ws + 16384 + 2359296);                   // kG*kE*4 = 384 KB
  float* out1   = (float*)(ws + 16384 + 2359296 + 393216);          // 384 KB

  hipLaunchKernelGGL(gtp_setup, dim3(1), dim3(kE), 0, stream,
                     query, Wq, bq, Wk, sizes, offsets, qkT);
  hipLaunchKernelGGL(gtp_attn, dim3(kG), dim3(128), 0, stream,
                     node_feat, sizes, offsets, qkT, xbar);
  hipLaunchKernelGGL(gtp_wv, dim3(kG), dim3(kE), 0, stream,
                     Wv, bv, xbar, pooled);
  hipLaunchKernelGGL(gtp_gemm, dim3(kE / 16, kG / 16), dim3(32), 0, stream,
                     pooled, Wo, bo, out1, kE, kE);
  hipLaunchKernelGGL(gtp_gemm, dim3(kOUT / 16, kG / 16), dim3(32), 0, stream,
                     out1, Wp, bp, out, kOUT, kE);
}
